// BlockSparseDilatedAttention_76828374990947
// MI455X (gfx1250) — compile-verified, matrix operand order checked by reading, not run on hardware
//
#include <hip/hip_runtime.h>
#include <hip/hip_bf16.h>

typedef __attribute__((ext_vector_type(16))) _Float16 v16h;
typedef __attribute__((ext_vector_type(4)))  _Float16 v4h;
typedef __attribute__((ext_vector_type(8)))  float    v8f;

// q,k,v,out layout: [S=8192][H=12][D=64] fp32, stride per position = 768 floats.
#define HD 768
#define SCALE 0.125f  // 1/sqrt(64)

__global__ void zero_fill_kernel(float* __restrict__ p, int n) {
    int i = blockIdx.x * blockDim.x + threadIdx.x;
    if (i < n) p[i] = 0.0f;
}

// One workgroup = (row block r, head, sub-block n). WAVES = L/16 waves; each
// wave owns a 16-row output tile and loops over the 3..5 col blocks in r's
// window, accumulating o += softmax(Q K^T * s) V per col block (independent
// softmax per col block, summed — matches the reference's at[].add()).
template<int L, int SEG, int DIL, int OFF, int NSUB>
__global__ __launch_bounds__((L / 16) * 32)
void attn_kernel(const float* __restrict__ q, const float* __restrict__ k,
                 const float* __restrict__ v, float* __restrict__ out, int h0) {
    constexpr int WAVES = L / 16;
    constexpr int NST   = L / 16;   // 16-wide key tiles per score stripe
    constexpr int T     = WAVES * 32;

    __shared__ _Float16 Klds[L * 64];
    __shared__ _Float16 Vlds[L * 64];
    __shared__ _Float16 Plds[WAVES * 16 * L];  // per-wave P scratch (repack D->A layout)

    const int bid  = blockIdx.x;
    const int n    = bid % NSUB;
    const int hh   = (bid / NSUB) % 4;
    const int r    = bid / (NSUB * 4);
    const int head = h0 + hh;

    const int tid   = threadIdx.x;
    const int wave  = tid >> 5;
    const int lane  = tid & 31;
    const int lhalf = (lane >> 4) & 1;  // which 16-lane half
    const int ln    = lane & 15;

    const int rowbase = r * 512 + n * SEG + OFF;  // query positions: rowbase + j*DIL

    // ---- Load Q A-fragments straight from global (rows = wave*16 + ln). ----
    // 16-bit A 16x32 layout: lane M = lane&15; element e -> K = (e&8)*2 + lhalf*8 + (e&7).
    // Per half-fragment the K run is 8 contiguous d values -> two aligned float4s.
    v16h aQ[2];
    {
        const int j = wave * 16 + ln;
        const float* qrow = q + (size_t)(rowbase + j * DIL) * HD + head * 64;
#pragma unroll
        for (int f = 0; f < 2; ++f) {
            const int dbase = f * 32 + (lhalf << 3);
            float4 x0 = *(const float4*)&qrow[dbase + 0];
            float4 x1 = *(const float4*)&qrow[dbase + 4];
            float4 x2 = *(const float4*)&qrow[dbase + 16];
            float4 x3 = *(const float4*)&qrow[dbase + 20];
            aQ[f][0]  = (_Float16)x0.x; aQ[f][1]  = (_Float16)x0.y;
            aQ[f][2]  = (_Float16)x0.z; aQ[f][3]  = (_Float16)x0.w;
            aQ[f][4]  = (_Float16)x1.x; aQ[f][5]  = (_Float16)x1.y;
            aQ[f][6]  = (_Float16)x1.z; aQ[f][7]  = (_Float16)x1.w;
            aQ[f][8]  = (_Float16)x2.x; aQ[f][9]  = (_Float16)x2.y;
            aQ[f][10] = (_Float16)x2.z; aQ[f][11] = (_Float16)x2.w;
            aQ[f][12] = (_Float16)x3.x; aQ[f][13] = (_Float16)x3.y;
            aQ[f][14] = (_Float16)x3.z; aQ[f][15] = (_Float16)x3.w;
        }
    }

    v8f o[4];
#pragma unroll
    for (int i = 0; i < 4; ++i) o[i] = (v8f){};

    const int c0 = (r - 2 < 0) ? 0 : r - 2;
    const int c1 = (r + 3 > 16) ? 16 : r + 3;

    for (int c = c0; c < c1; ++c) {
        __syncthreads();  // previous iteration's LDS reads are done
        // ---- Stage dilated K/V tile of col block c into LDS as f16 (float4 -> v4h). ----
        {
            const int colbase = c * 512 + n * SEG + OFF;
            constexpr int NV4 = L * 16;  // float4 chunks per matrix
            for (int idx = tid; idx < NV4; idx += T) {
                int row = idx >> 4;
                int dc  = (idx & 15) << 2;
                size_t g = (size_t)(colbase + row * DIL) * HD + head * 64 + dc;
                float4 kf = *(const float4*)&k[g];
                float4 vf = *(const float4*)&v[g];
                v4h kh = {(_Float16)kf.x, (_Float16)kf.y, (_Float16)kf.z, (_Float16)kf.w};
                v4h vh = {(_Float16)vf.x, (_Float16)vf.y, (_Float16)vf.z, (_Float16)vf.w};
                *(v4h*)&Klds[row * 64 + dc] = kh;
                *(v4h*)&Vlds[row * 64 + dc] = vh;
            }
            if (c + 1 < c1) {  // warm next col block (global_prefetch_b8)
                const int nb = (c + 1) * 512 + n * SEG + OFF;
                for (int row = tid; row < L; row += T) {
                    __builtin_prefetch(&k[(size_t)(nb + row * DIL) * HD + head * 64], 0, 3);
                    __builtin_prefetch(&v[(size_t)(nb + row * DIL) * HD + head * 64], 0, 3);
                }
            }
        }
        __syncthreads();

        // ---- Scores: 16 rows x L keys, D=64 -> two K=32 WMMA steps per tile. ----
        v8f sc[NST];
#pragma unroll
        for (int t = 0; t < NST; ++t) {
            v8f acc = (v8f){};
#pragma unroll
            for (int f = 0; f < 2; ++f) {
                v16h bK;
                const _Float16* krow = &Klds[(t * 16 + ln) * 64];
#pragma unroll
                for (int e = 0; e < 16; ++e) {
                    int d = f * 32 + ((e & 8) << 1) + (lhalf << 3) + (e & 7);
                    bK[e] = krow[d];
                }
                acc = __builtin_amdgcn_wmma_f32_16x16x32_f16(
                    false, aQ[f], false, bK, (short)0, acc, false, false);
            }
            sc[t] = acc;
        }

        // ---- Row softmax. C/D layout: element i -> row M = i + lhalf*8, col = ln.
        // Reductions over the 16 key-lanes stay inside each half (masks 1,2,4,8).
        float rinv[8];
#pragma unroll
        for (int i = 0; i < 8; ++i) {
            float m = -1e30f;
#pragma unroll
            for (int t = 0; t < NST; ++t) {
                sc[t][i] *= SCALE;
                m = fmaxf(m, sc[t][i]);
            }
#pragma unroll
            for (int msk = 1; msk < 16; msk <<= 1) m = fmaxf(m, __shfl_xor(m, msk));
            float su = 0.0f;
#pragma unroll
            for (int t = 0; t < NST; ++t) {
                float p = __expf(sc[t][i] - m);
                sc[t][i] = p;
                su += p;
            }
#pragma unroll
            for (int msk = 1; msk < 16; msk <<= 1) su += __shfl_xor(su, msk);
            rinv[i] = 1.0f / su;
        }

        // ---- Repack normalized P (D-layout) into per-wave LDS for A-fragment reads.
        _Float16* P = &Plds[wave * 16 * L];
#pragma unroll
        for (int t = 0; t < NST; ++t)
#pragma unroll
            for (int i = 0; i < 8; ++i) {
                int row = i + (lhalf << 3);
                int key = t * 16 + ln;
                P[row * L + key] = (_Float16)(sc[t][i] * rinv[i]);
            }
        // Per-wave LDS RAW: same-wave DS ops are in-order; wait out our own stores
        // instead of a full workgroup barrier (split-counter wait, CDNA5).
        asm volatile("s_wait_dscnt 0x0" ::: "memory");

        // ---- O += P (16xL) * V (Lx64): L/32 K-steps, 4 N-tiles of 16. ----
#pragma unroll
        for (int kt = 0; kt < L / 32; ++kt) {
            v16h aP;
#pragma unroll
            for (int e = 0; e < 16; ++e) {
                int key = kt * 32 + ((e & 8) << 1) + (lhalf << 3) + (e & 7);
                aP[e] = P[ln * L + key];
            }
#pragma unroll
            for (int nt = 0; nt < 4; ++nt) {
                v16h bV;
#pragma unroll
                for (int e = 0; e < 16; ++e) {
                    int key = kt * 32 + ((e & 8) << 1) + (lhalf << 3) + (e & 7);
                    bV[e] = Vlds[key * 64 + nt * 16 + ln];
                }
                o[nt] = __builtin_amdgcn_wmma_f32_16x16x32_f16(
                    false, aP, false, bV, (short)0, o[nt], false, false);
            }
        }
    }

    // ---- Write the 16x64 output tile (each (pos, head) owned by exactly one wave).
#pragma unroll
    for (int nt = 0; nt < 4; ++nt)
#pragma unroll
        for (int i = 0; i < 8; ++i) {
            int j = wave * 16 + i + (lhalf << 3);
            int d = nt * 16 + ln;
            size_t g = (size_t)(rowbase + j * DIL) * HD + head * 64 + d;
            out[g] = o[nt][i];
        }
}

extern "C" void kernel_launch(void* const* d_in, const int* in_sizes, int n_in,
                              void* d_out, int out_size, void* d_ws, size_t ws_size,
                              hipStream_t stream) {
    const float* q = (const float*)d_in[0];
    const float* k = (const float*)d_in[1];
    const float* v = (const float*)d_in[2];
    float* out = (float*)d_out;

    // Dilation leaves holes (heads 4-11): zero the output first.
    zero_fill_kernel<<<(out_size + 255) / 256, 256, 0, stream>>>(out, out_size);

    // Group 0: seg=128, dil=1, off=0, 4 sub-blocks, heads 0-3, L=128 (8 waves).
    attn_kernel<128, 128, 1, 0, 4><<<16 * 4 * 4, 256, 0, stream>>>(q, k, v, out, 0);
    // Group 1: seg=256, dil=2, off=1, 2 sub-blocks, heads 4-7, L=128 (8 waves).
    attn_kernel<128, 256, 2, 1, 2><<<16 * 4 * 2, 256, 0, stream>>>(q, k, v, out, 4);
    // Group 2: seg=128, dil=4, off=2, 4 sub-blocks, heads 8-11, L=32 (2 waves).
    attn_kernel<32, 128, 4, 2, 4><<<16 * 4 * 4, 64, 0, stream>>>(q, k, v, out, 8);
}
